// AttentionHead_73151882985755
// MI455X (gfx1250) — compile-verified, matrix operand order checked by reading, not run on hardware
//
#include <hip/hip_runtime.h>
#include <hip/hip_bf16.h>
#include <stdint.h>

typedef __attribute__((ext_vector_type(16))) __bf16       v16bf;
typedef __attribute__((ext_vector_type(8)))  __bf16       v8bf;
typedef __attribute__((ext_vector_type(8)))  float        v8f;
typedef __attribute__((ext_vector_type(4)))  float        v4f;
typedef __attribute__((ext_vector_type(8)))  unsigned int v8u;
typedef __attribute__((ext_vector_type(4)))  unsigned int u32x4;
typedef __attribute__((ext_vector_type(8)))  int          i32x8;
typedef __attribute__((ext_vector_type(4)))  int          i32x4;

#define WMMA_BF16(a, b, c) \
  __builtin_amdgcn_wmma_f32_16x16x32_bf16(false, (a), false, (b), (short)0, (c), false, false)

static constexpr int Bn = 8, Tn = 2048, Cn = 1024, Hn = 64;

// --- weight staging geometry (proj kernel) ---------------------------------
static constexpr int KCHUNK    = 256;                  // k-columns per chunk
static constexpr int NCHUNK    = Cn / KCHUNK;          // 4 chunks
static constexpr int WROWS     = 3 * Hn;               // 192 combined rows
static constexpr int ROWPITCH  = KCHUNK * 2 + 16;      // 528 B (16B TDM pad)
static constexpr int CHBYTES   = WROWS * ROWPITCH;     // 101376 B per buffer

static __device__ __forceinline__ unsigned int pack2bf(float lo, float hi) {
  unsigned short l = __builtin_bit_cast(unsigned short, (__bf16)lo);
  unsigned short h = __builtin_bit_cast(unsigned short, (__bf16)hi);
  return (unsigned int)l | ((unsigned int)h << 16);
}

// Issue one TDM load: chunk of wT (192 rows x 256 bf16, row stride 1024) into
// LDS at ldsoff, with 16B padding per 512B row (row pitch 528B).
static __device__ __forceinline__ void tdm_load_chunk(const __bf16* wT,
                                                      int chunk,
                                                      unsigned int ldsoff) {
  unsigned long long ga =
      (unsigned long long)(uintptr_t)wT + (unsigned long long)chunk * KCHUNK * 2;
  u32x4 g0;
  g0[0] = 1u;                                    // count=1 (valid descriptor)
  g0[1] = ldsoff;                                // lds_addr
  g0[2] = (unsigned int)ga;                      // global_addr[31:0]
  g0[3] = (unsigned int)((ga >> 32) & 0x1FFFFFFu) | (2u << 30);  // type=2
  i32x8 g1;
  g1[0] = (int)((1u << 16)       // data_size = 2 bytes
                | (1u << 20)     // pad_enable
                | (6u << 22)     // pad_interval: 128 dwords (512B rows)
                | (3u << 25));   // pad_amount: 4 dwords (16B)
  g1[1] = (int)(1024u << 16);    // tensor_dim0 = 1024 (low16 at [63:48])
  g1[2] = (int)(192u << 16);     // tensor_dim0 hi=0, tensor_dim1 = 192
  g1[3] = (int)(256u << 16);     // tensor_dim1 hi=0, tile_dim0 = 256
  g1[4] = (int)192u;             // tile_dim1 = 192, tile_dim2 = 0
  g1[5] = (int)1024;             // tensor_dim0_stride = 1024 (low32)
  g1[6] = 0;
  g1[7] = 0;
  i32x4 gz = {0, 0, 0, 0};
#if defined(__clang_major__) && __clang_major__ >= 23
  i32x8 gz8 = {0, 0, 0, 0, 0, 0, 0, 0};
  __builtin_amdgcn_tensor_load_to_lds(g0, g1, gz, gz, gz8, 0);
#else
  __builtin_amdgcn_tensor_load_to_lds(g0, g1, gz, gz, 0);
#endif
}

// ---------------------------------------------------------------------------
// Kernel 0: convert fp32 weights (C,H) to bf16, transposed -> wT[3][H][C]
// ---------------------------------------------------------------------------
__global__ __launch_bounds__(256) void wtrans_kernel(
    const float* __restrict__ wq, const float* __restrict__ wk,
    const float* __restrict__ wv, __bf16* __restrict__ wT) {
  int idx = blockIdx.x * blockDim.x + threadIdx.x;   // 3*64*1024 threads
  int m   = idx >> 16;
  int rem = idx & 65535;
  int n   = rem >> 10;
  int c   = rem & 1023;
  const float* w = (m == 0) ? wq : ((m == 1) ? wk : wv);
  wT[idx] = (__bf16)w[c * Hn + n];
}

// ---------------------------------------------------------------------------
// Kernel 1: fused QKV projection. One wave = 16 rows of x, all 64 cols, all
// 3 matrices (12 accumulators). x read once from HBM with one-step-ahead
// prefetch. Weights staged to LDS by the Tensor Data Mover, double-buffered,
// chunk c+1 DMA overlapping chunk c compute (TENSORcnt pipelining).
//   q,k : bf16 [B*T, 64] row-major       vT : bf16 [B][64][T]
// ---------------------------------------------------------------------------
__global__ __launch_bounds__(256) void proj_kernel(
    const float* __restrict__ x, const __bf16* __restrict__ wT,
    __bf16* __restrict__ q, __bf16* __restrict__ k, __bf16* __restrict__ vT) {
  __shared__ alignas(16) char smem[2 * CHBYTES];   // ~198 KB of 320 KB WGP LDS

  const int lane  = threadIdx.x & 31;
  const int l16   = lane & 15;
  const int lhalf = lane >> 4;
  const int wid   = threadIdx.x >> 5;
  const int wave  = blockIdx.x * (blockDim.x >> 5) + wid;
  const int m0    = wave * 16;   // global row tile

  v8f acc[12];
#pragma unroll
  for (int i = 0; i < 12; ++i) acc[i] = {};

  const float* xrow = x + (size_t)(m0 + l16) * Cn + lhalf * 16;

  // wave 0 kicks off the first two weight-chunk DMAs
  if (wid == 0) {
    tdm_load_chunk(wT, 0, (unsigned int)(uintptr_t)&smem[0]);
    tdm_load_chunk(wT, 1, (unsigned int)(uintptr_t)&smem[CHBYTES]);
  }

  // preload A raw for kk = 0
  v4f xr0 = *(const v4f*)(xrow + 0);
  v4f xr1 = *(const v4f*)(xrow + 4);
  v4f xr2 = *(const v4f*)(xrow + 8);
  v4f xr3 = *(const v4f*)(xrow + 12);

#pragma unroll
  for (int c = 0; c < NCHUNK; ++c) {
    if (wid == 0) {
      if (c < NCHUNK - 1) __builtin_amdgcn_s_wait_tensorcnt(1);
      else                __builtin_amdgcn_s_wait_tensorcnt(0);
    }
    __syncthreads();
    asm volatile("" ::: "memory");   // LDS written by TDM, not visible in IR
    const char* lb = smem + (c & 1) * CHBYTES;

    for (int ks = 0; ks < KCHUNK / 32; ++ks) {
      const int kk  = c * KCHUNK + ks * 32;
      const int kkn = (kk + 32 < Cn) ? kk + 32 : kk;
      // issue next A raw loads early
      v4f y0 = *(const v4f*)(xrow + kkn + 0);
      v4f y1 = *(const v4f*)(xrow + kkn + 4);
      v4f y2 = *(const v4f*)(xrow + kkn + 8);
      v4f y3 = *(const v4f*)(xrow + kkn + 12);
      // convert current A raw -> bf16 fragment
      v16bf a;
#pragma unroll
      for (int j = 0; j < 4; ++j) {
        a[j]      = (__bf16)xr0[j];
        a[4 + j]  = (__bf16)xr1[j];
        a[8 + j]  = (__bf16)xr2[j];
        a[12 + j] = (__bf16)xr3[j];
      }
      // 12 WMMAs, B fragments from LDS (row pitch 528B kills bank conflicts)
#pragma unroll
      for (int idx = 0; idx < 12; ++idx) {
        const char* bp =
            lb + (idx * 16 + l16) * ROWPITCH + (ks * 32 + lhalf * 16) * 2;
        v8bf blo = *(const v8bf*)(bp);
        v8bf bhi = *(const v8bf*)(bp + 16);
        v16bf bf;
#pragma unroll
        for (int j = 0; j < 8; ++j) { bf[j] = blo[j]; bf[8 + j] = bhi[j]; }
        acc[idx] = WMMA_BF16(a, bf, acc[idx]);
      }
      xr0 = y0; xr1 = y1; xr2 = y2; xr3 = y3;
    }

    __syncthreads();                 // all waves done with buffer (c&1)
    if (wid == 0 && c + 2 < NCHUNK)  // refill it two chunks ahead
      tdm_load_chunk(wT, c + 2,
                     (unsigned int)(uintptr_t)&smem[(c & 1) * CHBYTES]);
  }

  // C layout: lane<16 holds rows m0..m0+7 of column nt*16+l16; lane>=16 rows +8
  const int rowbase = m0 + lhalf * 8;
  const int b    = m0 >> 11;        // T = 2048
  const int tloc = m0 & (Tn - 1);
#pragma unroll
  for (int nt = 0; nt < 4; ++nt) {
    const int n = nt * 16 + l16;
#pragma unroll
    for (int r = 0; r < 8; ++r) {
      q[(size_t)(rowbase + r) * Hn + n] = (__bf16)acc[0 * 4 + nt][r];
      k[(size_t)(rowbase + r) * Hn + n] = (__bf16)acc[1 * 4 + nt][r];
    }
    v8bf vv;
#pragma unroll
    for (int r = 0; r < 8; ++r) vv[r] = (__bf16)acc[2 * 4 + nt][r];
    *(v8bf*)(vT + ((size_t)b * Hn + n) * Tn + tloc + lhalf * 8) = vv;
  }
}

// ---------------------------------------------------------------------------
// Kernel 2: causal flash attention, transposed tiles.
//   S^T = K_tile(16s x 64h) . Q^T(64h x 16t)   -> lane owns one t column
//   O^T = V^T(16h x 32s) . P^T(32s x 16t)      accumulated over s blocks
// K fragments prefetched one block ahead; vT loads hoisted above softmax.
// ---------------------------------------------------------------------------
__global__ __launch_bounds__(256) void attn_kernel(
    const __bf16* __restrict__ q, const __bf16* __restrict__ k,
    const __bf16* __restrict__ vT, float* __restrict__ out) {
  const int lane  = threadIdx.x & 31;
  const int l16   = lane & 15;
  const int lhalf = lane >> 4;
  const int wave  = blockIdx.x * (blockDim.x >> 5) + (threadIdx.x >> 5);
  const int b     = wave >> 7;                 // 128 t-tiles per batch
  const int t0    = (127 - (wave & 127)) * 16; // heavy tiles first
  const float scale = 0.03125f;                // C^-0.5 = 1/32

  // Q^T B-fragments for both h k-steps, held across the whole s loop
  v16bf qB[2];
#pragma unroll
  for (int i = 0; i < 2; ++i)
    qB[i] = *(const v16bf*)(q + (size_t)(b * Tn + t0 + l16) * Hn + i * 32 +
                            lhalf * 16);

  v8f O[4];
#pragma unroll
  for (int i = 0; i < 4; ++i) O[i] = {};
  float m_run = -1e30f, l_run = 0.f;

  const int t  = t0 + l16;
  const int nb = (t0 + 47) >> 5;         // causal: s blocks of 32 up to t0+15

  // preload K fragments for block 0:  kc[h0*2+0]=rows s0..15, +1=rows s0+16..31
  const __bf16* kb = k + (size_t)(b * Tn) * Hn + lhalf * 16;
  v16bf kc[4];
#pragma unroll
  for (int h0 = 0; h0 < 2; ++h0) {
    kc[h0 * 2 + 0] = *(const v16bf*)(kb + (size_t)(l16) * Hn + h0 * 32);
    kc[h0 * 2 + 1] = *(const v16bf*)(kb + (size_t)(16 + l16) * Hn + h0 * 32);
  }

  for (int blk = 0; blk < nb; ++blk) {
    const int s0 = blk * 32;
    // ---- vT fragments for this block: consumed only after softmax ----
    v16bf vt[4];
#pragma unroll
    for (int ht = 0; ht < 4; ++ht)
      vt[ht] = *(const v16bf*)(vT + (size_t)(b * Hn + ht * 16 + l16) * Tn +
                               s0 + lhalf * 16);
    // ---- scores S^T: two 16x16 tiles (s0..15, s0+16..31) ----
    v8f Sa = {}, Sb = {};
#pragma unroll
    for (int h0 = 0; h0 < 2; ++h0) {
      Sa = WMMA_BF16(kc[h0 * 2 + 0], qB[h0], Sa);
      Sb = WMMA_BF16(kc[h0 * 2 + 1], qB[h0], Sb);
    }
    // ---- prefetch next block's K fragments (clamped, wave-uniform) ----
    const int s0n = (s0 + 32 < Tn - 32) ? s0 + 32 : s0;
    v16bf kn[4];
#pragma unroll
    for (int h0 = 0; h0 < 2; ++h0) {
      kn[h0 * 2 + 0] =
          *(const v16bf*)(kb + (size_t)(s0n + l16) * Hn + h0 * 32);
      kn[h0 * 2 + 1] =
          *(const v16bf*)(kb + (size_t)(s0n + 16 + l16) * Hn + h0 * 32);
    }
    // ---- scale + causal mask (lane owns column t; rows s in VGPRs) ----
    float va[8], vb[8];
    float mloc = -1e30f;
#pragma unroll
    for (int r = 0; r < 8; ++r) {
      int sA = s0 + r + lhalf * 8;
      int sB = sA + 16;
      va[r] = (sA <= t) ? Sa[r] * scale : -1e30f;
      vb[r] = (sB <= t) ? Sb[r] * scale : -1e30f;
      mloc  = fmaxf(mloc, fmaxf(va[r], vb[r]));
    }
    mloc = fmaxf(mloc, __shfl_xor(mloc, 16, 32));       // combine lane pair
    float m_new = fmaxf(m_run, mloc);
    float alpha = __expf(m_run - m_new);
    // ---- exponentiate, pack to bf16 pairs, partial sum ----
    float sloc = 0.f;
    unsigned int pa[4], pb[4];
#pragma unroll
    for (int j = 0; j < 4; ++j) {
      float e0 = __expf(va[2 * j] - m_new);
      float e1 = __expf(va[2 * j + 1] - m_new);
      float f0 = __expf(vb[2 * j] - m_new);
      float f1 = __expf(vb[2 * j + 1] - m_new);
      sloc += (e0 + e1) + (f0 + f1);
      pa[j] = pack2bf(e0, e1);
      pb[j] = pack2bf(f0, f1);
    }
    l_run = l_run * alpha + (sloc + __shfl_xor(sloc, 16, 32));
    m_run = m_new;
    // ---- rescale running O by alpha ----
#pragma unroll
    for (int i = 0; i < 4; ++i)
#pragma unroll
      for (int r = 0; r < 8; ++r) O[i][r] *= alpha;
    // ---- half-swap between lane pairs to form P^T B-fragment (K=32) ----
    unsigned int pf[8];
#pragma unroll
    for (int j = 0; j < 4; ++j) {
      unsigned int send = lhalf ? pa[j] : pb[j];
      unsigned int recv = __shfl_xor(send, 16, 32);
      pf[j]     = lhalf ? recv  : pa[j];   // K 0..7  (lane<16) / 16..23
      pf[4 + j] = lhalf ? pb[j] : recv;    // K 8..15 recv / 24..31 own
    }
    v8u pu;
#pragma unroll
    for (int j = 0; j < 8; ++j) pu[j] = pf[j];
    v16bf pB = __builtin_bit_cast(v16bf, pu);
    // ---- O^T += V^T . P^T over the 4 h tiles ----
#pragma unroll
    for (int ht = 0; ht < 4; ++ht) O[ht] = WMMA_BF16(vt[ht], pB, O[ht]);
    // rotate K prefetch buffer
#pragma unroll
    for (int i = 0; i < 4; ++i) kc[i] = kn[i];
  }

  // ---- normalize and store out[b, t, h] (fp32, b128 stores) ----
  float inv = 1.f / l_run;
#pragma unroll
  for (int ht = 0; ht < 4; ++ht) {
    v4f o0, o1;
#pragma unroll
    for (int r = 0; r < 4; ++r) {
      o0[r] = O[ht][r] * inv;
      o1[r] = O[ht][4 + r] * inv;
    }
    float* op = out + (size_t)(b * Tn + t0 + l16) * Hn + ht * 16 + lhalf * 8;
    *(v4f*)op       = o0;
    *(v4f*)(op + 4) = o1;
  }
}

// ---------------------------------------------------------------------------
extern "C" void kernel_launch(void* const* d_in, const int* in_sizes, int n_in,
                              void* d_out, int out_size, void* d_ws,
                              size_t ws_size, hipStream_t stream) {
  const float* x  = (const float*)d_in[0];
  const float* wq = (const float*)d_in[1];
  const float* wk = (const float*)d_in[2];
  const float* wv = (const float*)d_in[3];
  float* out = (float*)d_out;

  char* ws = (char*)d_ws;
  __bf16* q  = (__bf16*)(ws);                         // 2 MB  [B*T, 64]
  __bf16* k  = (__bf16*)(ws + ((size_t)2 << 20));     // 2 MB  [B*T, 64]
  __bf16* vT = (__bf16*)(ws + ((size_t)4 << 20));     // 2 MB  [B][64][T]
  __bf16* wT = (__bf16*)(ws + ((size_t)6 << 20));     // 384 KB [3][64][1024]

  hipLaunchKernelGGL(wtrans_kernel, dim3(768), dim3(256), 0, stream,
                     wq, wk, wv, wT);
  hipLaunchKernelGGL(proj_kernel, dim3(128), dim3(256), 0, stream,
                     x, wT, q, k, vT);
  hipLaunchKernelGGL(attn_kernel, dim3(128), dim3(256), 0, stream,
                     q, k, vT, out);
}
